// PointerNet_53102975648360
// MI455X (gfx1250) — matrix operand order repaired and checked by reading, not compiled, not run on hardware
//
#include <hip/hip_runtime.h>

// ---------------------------------------------------------------------------
// PointerNet rela_pred fused kernels for gfx1250 (MI455X, wave32, WMMA).
//
// Shapes: B=4, N=128, H=512, 2H=1024, d_pair=256, out channels = 3*258 = 774.
//
//   stage 1: U = P @ W1[:512],  V = P @ W1[512:]     (f32, tiny, -> ws)
//            W2 -> bf16, transposed [n][k]           (-> ws, WMMA B layout)
//   stage 2: per (mlp, b, i, j-tile64):
//            A = bf16(relu(U[i,:] + V[j,:]))  built in LDS (64 x 1024)
//            C = A @ W2  via v_wmma_f32_16x16x32_bf16, f32 accum
//            relu -> store 256 channels; w3 head -> 2-way softmax -> 2 ch.
//
// M_TILE = 64 pairs/block halves W2 L2 traffic vs a 32-pair tile (each block
// streams the whole 512 KB bf16 W2 once per K sweep regardless of M).
// ---------------------------------------------------------------------------

typedef __attribute__((ext_vector_type(8)))  float  v8f;
typedef __attribute__((ext_vector_type(16))) __bf16 v16bf;

struct Frag32 { uint4 lo, hi; };

static __device__ __forceinline__ v16bf make_frag(uint4 lo, uint4 hi) {
    Frag32 f{lo, hi};
    return __builtin_bit_cast(v16bf, f);
}

static __device__ __forceinline__ unsigned short f32_to_bf16_bits(float x) {
    return __builtin_bit_cast(unsigned short, (__bf16)x);
}

static __device__ __forceinline__ unsigned int pack_bf16x2(float a, float b) {
    return (unsigned int)f32_to_bf16_bits(a) | ((unsigned int)f32_to_bf16_bits(b) << 16);
}

// ---------------------------------------------------------------------------
// Kernel 1: W2 (f32 [1024][256]) -> bf16 W2^T [mlp][256][1024] in workspace.
// Column-major-in-K so a B fragment (lane n = lane%16, K-run = (lane/16)*16)
// is a single contiguous 32-byte run -> two global b128 loads per fragment.
// ---------------------------------------------------------------------------
__global__ __launch_bounds__(256) void k_w2t(const float* __restrict__ W2f,
                                             const float* __restrict__ W2h1,
                                             const float* __restrict__ W2h2,
                                             unsigned short* __restrict__ w2t) {
    int blk = blockIdx.x;           // 0..767
    int mlp = blk >> 8;
    int n   = blk & 255;
    const float* W2 = (mlp == 0) ? W2f : (mlp == 1 ? W2h1 : W2h2);
    unsigned short* dst = w2t + ((size_t)mlp * 256 + n) * 1024;
    for (int k = threadIdx.x; k < 1024; k += 256)
        dst[k] = f32_to_bf16_bits(W2[(size_t)k * 256 + n]);
}

// ---------------------------------------------------------------------------
// Kernel 2: U/V = P @ W1 halves, f32 accuracy.  6 matrices (3 mlp x {U,V}),
// each [512 rows][1024 cols], K=512.  Block: 16 rows x 1024 cols, P tile in
// LDS (broadcast reads), W1 read as coalesced float4 (stays in L2).
// ---------------------------------------------------------------------------
__global__ __launch_bounds__(256) void k_uv(const float* __restrict__ P,
                                            const float* __restrict__ W1f,
                                            const float* __restrict__ W1h1,
                                            const float* __restrict__ W1h2,
                                            float* __restrict__ Uws,
                                            float* __restrict__ Vws) {
    int mat  = blockIdx.x >> 5;     // 0..5
    int rt   = blockIdx.x & 31;     // 32 row tiles of 16
    int mlp  = mat >> 1;
    int half = mat & 1;
    const float* W1    = (mlp == 0) ? W1f : (mlp == 1 ? W1h1 : W1h2);
    const float* Wbase = W1 + (size_t)half * 512 * 1024;
    float* dst = (half ? Vws : Uws) + (size_t)mlp * 512 * 1024 + (size_t)rt * 16 * 1024;
    const float* Pbase = P + (size_t)rt * 16 * 512;

    __shared__ float sP[16][512];
    for (int t = threadIdx.x; t < 16 * 512; t += 256)
        sP[t >> 9][t & 511] = Pbase[t];
    __syncthreads();

    const int c0 = threadIdx.x * 4;           // 4 consecutive output cols
    float acc[16][4];
#pragma unroll
    for (int r = 0; r < 16; ++r) { acc[r][0] = acc[r][1] = acc[r][2] = acc[r][3] = 0.f; }

    for (int k = 0; k < 512; ++k) {
        float4 w = *(const float4*)(Wbase + (size_t)k * 1024 + c0);
#pragma unroll
        for (int r = 0; r < 16; ++r) {
            float p = sP[r][k];
            acc[r][0] = fmaf(p, w.x, acc[r][0]);
            acc[r][1] = fmaf(p, w.y, acc[r][1]);
            acc[r][2] = fmaf(p, w.z, acc[r][2]);
            acc[r][3] = fmaf(p, w.w, acc[r][3]);
        }
    }
#pragma unroll
    for (int r = 0; r < 16; ++r) {
        float4 o{acc[r][0], acc[r][1], acc[r][2], acc[r][3]};
        *(float4*)(dst + (size_t)r * 1024 + c0) = o;
    }
}

// ---------------------------------------------------------------------------
// Kernel 3: fused pair-GEMM.  Block = (mlp, b, i, j0..j0+63), 256 threads.
//   Phase 1: A[m][k] = bf16(relu(U[b,i,k] + V[b,j0+m,k])) -> LDS (row pad 1032
//            elements so per-lane fragment ds_load_b128 hits distinct banks);
//            float4 global reads, packed-bf16 ds_store_b64 writes.
//   Phase 2: 8 waves; wave w owns cols [32w,32w+32).  4 M-subtiles x 2 N-tiles
//            accumulators; 32 K-chunks x 8 v_wmma_f32_16x16x32_bf16.
//   Phase 3: relu -> store; per-row w3 dot via shfl_xor reduce + ds_add_f32;
//            2-way softmax -> channels 256/257.
// ---------------------------------------------------------------------------
__global__ __launch_bounds__(256) void k_pair(const float* __restrict__ Uws,
                                              const float* __restrict__ Vws,
                                              const unsigned short* __restrict__ w2t,
                                              const float* __restrict__ w3f,
                                              const float* __restrict__ w3h1,
                                              const float* __restrict__ w3h2,
                                              float* __restrict__ out) {
    const int x   = blockIdx.x;                // 3072 blocks
    const int jt  = x & 1;                     // 2 j-tiles of 64
    const int i   = (x >> 1) & 127;
    const int b   = (x >> 8) & 3;
    const int mlp = x >> 10;                   // 0..2
    const float* w3p = (mlp == 0) ? w3f : (mlp == 1 ? w3h1 : w3h2);
    const int j0 = jt * 64;

    const float* Urow  = Uws + ((size_t)mlp * 512 + b * 128 + i) * 1024;
    const float* Vbase = Vws + ((size_t)mlp * 512 + b * 128 + j0) * 1024;
    const unsigned short* Wb = w2t + (size_t)mlp * 256 * 1024;

    __shared__ __align__(16) unsigned short As[64 * 1032];   // 132 KB, padded rows
    __shared__ float hacc[64][2];

    const int tid = threadIdx.x;
    if (tid < 128) ((float*)hacc)[tid] = 0.f;

    // ---- Phase 1: build bf16 A tile (64 pairs x 1024 K) in LDS ----
    // step s: all 256 threads work on row m=s, thread t covers k = 4t..4t+3.
    {
        const float4* U4 = (const float4*)Urow;
        float4 u = U4[tid & 255];              // same per step; reload cheap (L0)
#pragma unroll 2
        for (int m = 0; m < 64; ++m) {
            float4 v = *(const float4*)(Vbase + (size_t)m * 1024 + tid * 4);
            float z0 = fmaxf(u.x + v.x, 0.f);
            float z1 = fmaxf(u.y + v.y, 0.f);
            float z2 = fmaxf(u.z + v.z, 0.f);
            float z3 = fmaxf(u.w + v.w, 0.f);
            uint2 pk{pack_bf16x2(z0, z1), pack_bf16x2(z2, z3)};
            *(uint2*)(As + m * 1032 + tid * 4) = pk;
        }
    }
    __syncthreads();

    // ---- Phase 2: WMMA main loop ----
    const int lane = tid & 31, wave = tid >> 5;
    const int lmod = lane & 15, kg = lane >> 4;

    // A fragment (16x32 bf16): lane row = lmod (+subtile), contiguous 16-K run chosen by kg.
    const unsigned short* a0p = As + lmod * 1032 + kg * 16;
    // B fragment (32x16 bf16): lane col n, contiguous 16-K run chosen by kg.
    const int n_t0 = wave * 32 + lmod;
    const unsigned short* b0p = Wb + (size_t)n_t0 * 1024 + kg * 16;
    const unsigned short* b1p = b0p + 16 * 1024;

    v8f acc[4][2];
#pragma unroll
    for (int s = 0; s < 4; ++s) { acc[s][0] = (v8f){}; acc[s][1] = (v8f){}; }

#pragma unroll 2
    for (int c = 0; c < 32; ++c) {
        const int ko = c * 32;
        v16bf bf[2];
        bf[0] = make_frag(*(const uint4*)(b0p + ko), *(const uint4*)(b0p + ko + 8));
        bf[1] = make_frag(*(const uint4*)(b1p + ko), *(const uint4*)(b1p + ko + 8));
        v16bf af[4];
#pragma unroll
        for (int s = 0; s < 4; ++s) {
            const unsigned short* ap = a0p + s * (16 * 1032) + ko;
            af[s] = make_frag(*(const uint4*)ap, *(const uint4*)(ap + 8));
        }
#pragma unroll
        for (int s = 0; s < 4; ++s) {
#pragma unroll
            for (int t = 0; t < 2; ++t) {
                acc[s][t] = __builtin_amdgcn_wmma_f32_16x16x32_bf16(
                    false, af[s], false, bf[t], (short)0, acc[s][t], false, false);
            }
        }
    }

    // ---- Phase 3: relu, store, softmax head ----
    const int n_t1 = n_t0 + 16;
    const float w30_0 = w3p[2 * n_t0], w31_0 = w3p[2 * n_t0 + 1];
    const float w30_1 = w3p[2 * n_t1], w31_1 = w3p[2 * n_t1 + 1];
    float* obase = out + (((size_t)b * 128 + i) * 128 + j0) * 774 + (size_t)mlp * 258;

#pragma unroll
    for (int s = 0; s < 4; ++s) {
#pragma unroll
        for (int r = 0; r < 8; ++r) {
            int row = s * 16 + kg * 8 + r;       // C/D layout: lanes 16-31 -> M+8
            float v0 = fmaxf(acc[s][0][r], 0.f);
            float v1 = fmaxf(acc[s][1][r], 0.f);
            float* orow = obase + (size_t)row * 774;
            orow[n_t0] = v0;
            orow[n_t1] = v1;
            float p0 = v0 * w30_0 + v1 * w30_1;
            float p1 = v0 * w31_0 + v1 * w31_1;
#pragma unroll
            for (int m = 1; m < 16; m <<= 1) {   // reduce within each 16-lane half
                p0 += __shfl_xor(p0, m, 32);
                p1 += __shfl_xor(p1, m, 32);
            }
            if (lmod == 0) {
                atomicAdd(&hacc[row][0], p0);    // ds_add_f32
                atomicAdd(&hacc[row][1], p1);
            }
        }
    }

    __syncthreads();
    if (tid < 64) {
        float s0 = hacc[tid][0], s1 = hacc[tid][1];
        float mx = fmaxf(s0, s1);
        float e0 = __expf(s0 - mx), e1 = __expf(s1 - mx);
        float inv = 1.f / (e0 + e1);
        float* orow = obase + (size_t)tid * 774;
        orow[256] = e0 * inv;
        orow[257] = e1 * inv;
    }
}

// ---------------------------------------------------------------------------
// Workspace layout:
//   [0, 1.5MB)   : W2^T bf16   [3][256][1024]
//   [..,+6MB)    : U f32       [3][512][1024]
//   [..,+6MB)    : V f32       [3][512][1024]
// Total ~13.5 MB.
// ---------------------------------------------------------------------------
extern "C" void kernel_launch(void* const* d_in, const int* in_sizes, int n_in,
                              void* d_out, int out_size, void* d_ws, size_t ws_size,
                              hipStream_t stream) {
    (void)in_sizes; (void)n_in; (void)out_size; (void)ws_size;
    const float* P    = (const float*)d_in[0];
    const float* Wf1  = (const float*)d_in[1];
    const float* Wf2  = (const float*)d_in[2];
    const float* w3   = (const float*)d_in[3];
    const float* Wh1a = (const float*)d_in[4];
    const float* Wh1b = (const float*)d_in[5];
    const float* wl1  = (const float*)d_in[6];
    const float* Wh2a = (const float*)d_in[7];
    const float* Wh2b = (const float*)d_in[8];
    const float* wl2  = (const float*)d_in[9];

    char* wsb = (char*)d_ws;
    unsigned short* w2t = (unsigned short*)wsb;
    float* U = (float*)(wsb + (size_t)3 * 256 * 1024 * 2);
    float* V = U + (size_t)3 * 512 * 1024;

    k_w2t <<<768,  256, 0, stream>>>(Wf2, Wh1b, Wh2b, w2t);
    k_uv  <<<192,  256, 0, stream>>>(P, Wf1, Wh1a, Wh2a, U, V);
    k_pair<<<3072, 256, 0, stream>>>(U, V, w2t, w3, wl1, wl2, (float*)d_out);
}
